// DSV4MainKVProjection_53755810677000
// MI455X (gfx1250) — matrix operand order repaired and checked by reading, not compiled, not run on hardware
//
#include <hip/hip_runtime.h>
#include <stdint.h>

typedef __attribute__((ext_vector_type(16))) __bf16        v16bf;
typedef __attribute__((ext_vector_type(8)))  float         v8f;
typedef __attribute__((ext_vector_type(8)))  unsigned int  v8u;

#define S_LEN   4096
#define H_DIM   4096
#define D_DIM   512
#define M_TILE  64
#define KT_STEPS (H_DIM / 32)   // 128 K-steps of 32
#define NT_TILES (D_DIM / 16)   // 32 N-tiles of 16

// f32 pair -> packed bf16 (round-to-nearest, ties away): 2 adds + 1 v_perm
__device__ __forceinline__ uint32_t bfpack2(float lo, float hi) {
  uint32_t ulo = __float_as_uint(lo) + 0x8000u;
  uint32_t uhi = __float_as_uint(hi) + 0x8000u;
  // result = { uhi[31:16], ulo[31:16] }  (S0 = uhi, S1 = ulo)
  return __builtin_amdgcn_perm(uhi, ulo, 0x07060302u);
}

// ------------------------------------------------------------------
// Kernel 1: pack wkv_weight (D x H, f32 row-major) into bf16 WMMA-B
// fragment-ready layout: [kt][nt][lane][dword], lane's 8 dwords
// contiguous (32B). B k order used here (matches A): lanes 0-15 -> n, K=0..15 in
// V0..7 (sequential pairs); lanes 16-31 -> n, K=16..31.
// ------------------------------------------------------------------
__global__ __launch_bounds__(256)
void pack_weight(const float* __restrict__ w, uint32_t* __restrict__ wp) {
  const int idx  = blockIdx.x * 256 + threadIdx.x;   // 0 .. 1M-1 dwords
  const int j    = idx & 7;                          // dword within lane
  const int lane = (idx >> 3) & 31;
  const int nt   = (idx >> 8) & 31;
  const int kt   = idx >> 13;                        // 0..127
  const int n    = nt * 16 + (lane & 15);
  const int k    = kt * 32 + ((lane >> 4) << 4) + (j << 1);
  const float* p = w + (size_t)n * H_DIM + k;
  wp[idx] = bfpack2(p[0], p[1]);
}

// ------------------------------------------------------------------
// Kernel 2: fused  kv = x @ W^T  ->  RMSNorm -> (quant448 | rope64)
// Block: 256 thr (8 waves). Tile: 64 rows x full 512 cols.
// Wave w owns cols [w*64, w*64+64), rows 0..63 (4 M-frags x 4 N-frags).
// Epilogue runs in 2 passes of 32 rows through a reused 64 KB kv tile.
// ------------------------------------------------------------------
union SMem {
  uint16_t a[2][M_TILE][32];          // double-buffered bf16 A tile (8 KB)
  struct {
    float kv[32][D_DIM];              // 64 KB kv pass buffer
    float part[32][8];
    float rstd[32];
  } e;
};

__global__ __launch_bounds__(256)
void fused_kvproj(const float* __restrict__ x,
                  const uint32_t* __restrict__ wp,
                  const float* __restrict__ rmsw,
                  float* __restrict__ out) {
  __shared__ SMem sm;
  const int t    = threadIdx.x;
  const int lane = t & 31;
  const int wv   = t >> 5;
  const int ln   = lane & 15;
  const int hi   = lane >> 4;
  const size_t rowBase = (size_t)blockIdx.x * M_TILE;

  // A-staging: thread t covers row ar, cols [ac,ac+4) and [ac+16,ac+20)
  // -> each 4-lane group's clause load is 64B contiguous.
  const int ar = t >> 2;               // 0..63
  const int ac = (t & 3) << 2;         // 0,4,8,12
  const float* xrow = x + (rowBase + ar) * (size_t)H_DIM + ac;

  v8f acc[4][4];
#pragma unroll
  for (int mt = 0; mt < 4; ++mt)
#pragma unroll
    for (int nf = 0; nf < 4; ++nf)
      acc[mt][nf] = (v8f){0.f, 0.f, 0.f, 0.f, 0.f, 0.f, 0.f, 0.f};

  // prologue: stage K-tile 0
  float4 s0 = *(const float4*)xrow;
  float4 s1 = *(const float4*)(xrow + 16);
  *(uint2*)&sm.a[0][ar][ac] =
      make_uint2(bfpack2(s0.x, s0.y), bfpack2(s0.z, s0.w));
  *(uint2*)&sm.a[0][ar][ac + 16] =
      make_uint2(bfpack2(s1.x, s1.y), bfpack2(s1.z, s1.w));

  for (int kt = 0; kt < KT_STEPS; ++kt) {
    __syncthreads();                              // buf kt&1 ready
    if (kt + 1 < KT_STEPS) {                      // prefetch next A tile
      s0 = *(const float4*)(xrow + (size_t)(kt + 1) * 32);
      s1 = *(const float4*)(xrow + (size_t)(kt + 1) * 32 + 16);
    }

    const int buf = kt & 1;

    // A fragments from LDS (layout: V0-7 = K kb..kb+15, contiguous)
    v16bf afr[4];
#pragma unroll
    for (int mt = 0; mt < 4; ++mt) {
      const int m  = mt * 16 + ln;
      const int kb = hi * 16;
      uint4 l0 = *(const uint4*)&sm.a[buf][m][kb];
      uint4 l1 = *(const uint4*)&sm.a[buf][m][kb + 8];
      v8u au = {l0.x, l0.y, l0.z, l0.w, l1.x, l1.y, l1.z, l1.w};
      afr[mt] = __builtin_bit_cast(v16bf, au);
    }

    // B fragments straight from L2-resident packed weight + 16 WMMAs
#pragma unroll
    for (int nf = 0; nf < 4; ++nf) {
      const int nt = wv * 4 + nf;
      const uint32_t* bp =
          wp + ((size_t)(kt * NT_TILES + nt) * 32 + lane) * 8;
      uint4 b0 = *(const uint4*)bp;
      uint4 b1 = *(const uint4*)(bp + 4);
      v8u bu = {b0.x, b0.y, b0.z, b0.w, b1.x, b1.y, b1.z, b1.w};
      v16bf bfr = __builtin_bit_cast(v16bf, bu);
#pragma unroll
      for (int mt = 0; mt < 4; ++mt)
        acc[mt][nf] = __builtin_amdgcn_wmma_f32_16x16x32_bf16(
            false, afr[mt], false, bfr, (short)0, acc[mt][nf], false, false);
    }

    if (kt + 1 < KT_STEPS) {                      // fill other buffer
      *(uint2*)&sm.a[buf ^ 1][ar][ac] =
          make_uint2(bfpack2(s0.x, s0.y), bfpack2(s0.z, s0.w));
      *(uint2*)&sm.a[buf ^ 1][ar][ac + 16] =
          make_uint2(bfpack2(s1.x, s1.y), bfpack2(s1.z, s1.w));
    }
  }

  // ---------------- fused epilogue: 2 passes of 32 rows ----------------
  const int r  = t >> 3;   // 0..31
  const int ch = t & 7;    // 0..7

  for (int h = 0; h < 2; ++h) {
    __syncthreads();   // h==0: done with A buffers; h==1: done reading kv

    // C layout: VGPR p, lanes 0-15 -> M=p, N=lane; lanes 16-31 -> M=p+8
#pragma unroll
    for (int mt2 = 0; mt2 < 2; ++mt2)
#pragma unroll
      for (int nf = 0; nf < 4; ++nf) {
        const int n = wv * 64 + nf * 16 + ln;
        const v8f c = acc[h * 2 + mt2][nf];
#pragma unroll
        for (int p = 0; p < 8; ++p)
          sm.e.kv[mt2 * 16 + p + hi * 8][n] = c[p];
      }
    __syncthreads();

    // per-row sum of squares (8 partials per row)
    {
      float ss = 0.f;
#pragma unroll 8
      for (int j = 0; j < 64; ++j) {
        float v = sm.e.kv[r][ch * 64 + j];
        ss = fmaf(v, v, ss);
      }
      sm.e.part[r][ch] = ss;
    }
    __syncthreads();
    if (t < 32) {
      float s2 = 0.f;
#pragma unroll
      for (int j = 0; j < 8; ++j) s2 += sm.e.part[t][j];
      sm.e.rstd[t] = rsqrtf(s2 * (1.0f / (float)D_DIM) + 1e-6f);
    }
    __syncthreads();

    const float rstd = sm.e.rstd[r];
    const size_t gm  = rowBase + h * 32 + r;
    float* orow = out + gm * D_DIM;

    if (ch < 7) {
      // one 64-wide quant block per thread
      const int base = ch * 64;
      float amax = 1e-4f;
#pragma unroll 8
      for (int j = 0; j < 64; ++j) {
        float v = sm.e.kv[r][base + j] * rstd * rmsw[base + j];
        amax = fmaxf(amax, fabsf(v));
      }
      const float scale = amax * (1.0f / 448.0f);
      const float inv   = 448.0f / amax;
      const float step  = 448.0f / 127.0f;
#pragma unroll 8
      for (int j = 0; j < 64; ++j) {
        float v  = sm.e.kv[r][base + j] * rstd * rmsw[base + j];
        float xs = fminf(fmaxf(v * inv, -448.f), 448.f);
        orow[base + j] = rintf(xs * (127.0f / 448.0f)) * step * scale;
      }
    } else {
      // RoPE on last 64 dims
      const int s = (int)(gm & (S_LEN - 1));
#pragma unroll 4
      for (int i = 0; i < 32; ++i) {
        const int d0 = 448 + 2 * i;
        float a = sm.e.kv[r][d0]     * rstd * rmsw[d0];
        float b = sm.e.kv[r][d0 + 1] * rstd * rmsw[d0 + 1];
        // freq = 10000^(-i/32) ; ln(10000)/32 = 0.28782313662425572
        float freq = expf(-0.28782313662425572f * (float)i);
        float ang  = (float)s * freq;
        float sn, cs;
        sincosf(ang, &sn, &cs);
        orow[d0]     = a * cs - b * sn;
        orow[d0 + 1] = a * sn + b * cs;
      }
    }
  }
}

extern "C" void kernel_launch(void* const* d_in, const int* in_sizes, int n_in,
                              void* d_out, int out_size, void* d_ws, size_t ws_size,
                              hipStream_t stream) {
  const float* x    = (const float*)d_in[0];   // (4, 4096, 4096) f32
  const float* w    = (const float*)d_in[1];   // (512, 4096) f32
  const float* rmsw = (const float*)d_in[2];   // (512,) f32
  float*       out  = (float*)d_out;           // (4, 4096, 512) f32
  uint32_t*    wp   = (uint32_t*)d_ws;         // 4 MB packed bf16 weight

  // 1M packed dwords
  pack_weight<<<dim3(4096), dim3(256), 0, stream>>>(w, wp);
  // 16384 rows / 64 rows per block
  fused_kvproj<<<dim3(256), dim3(256), 0, stream>>>(x, wp, rmsw, out);
}
